// SelfAttention_10264971838239
// MI455X (gfx1250) — matrix operand order repaired and checked by reading, not compiled, not run on hardware
//
#include <hip/hip_runtime.h>
#include <hip/hip_bf16.h>

// ---------------------------------------------------------------------------
// CDNA5 (gfx1250) mixed-precision self-attention:
//   - v_wmma_f32_16x16x32_bf16 for all matmuls (fp32 accumulate)
//   - Tensor Data Mover (tensor_load_to_lds + s_wait_tensorcnt) stages K tiles
//   - transposed LDS tiles so every WMMA fragment read is a contiguous b128
// ---------------------------------------------------------------------------

typedef __attribute__((ext_vector_type(16))) __bf16          v16bf;
typedef __attribute__((ext_vector_type(16))) unsigned short  v16us;
typedef __attribute__((ext_vector_type(8)))  float           v8f;
typedef __attribute__((ext_vector_type(4)))  unsigned int    v4u;
typedef __attribute__((ext_vector_type(8)))  unsigned int    v8u;

#define USE_TDM 1

__device__ __forceinline__ unsigned short f2bf(float f) {
    unsigned u = __float_as_uint(f);
    unsigned r = u + 0x7fffu + ((u >> 16) & 1u);   // round-to-nearest-even
    return (unsigned short)(r >> 16);
}
__device__ __forceinline__ float bf2f(unsigned short u) {
    return __uint_as_float(((unsigned)u) << 16);
}
__device__ __forceinline__ v8f wmma_bf16(v16us a, v16us b, v8f c) {
    return __builtin_amdgcn_wmma_f32_16x16x32_bf16(
        false, __builtin_bit_cast(v16bf, a),
        false, __builtin_bit_cast(v16bf, b),
        (short)0, c, false, false);
}

#if USE_TDM
// TDM: load a 2-D tile (32 rows x 64 ushort) from global into LDS.
// Row stride in global = row_stride_elems ushorts. LDS rows are padded
// 64+8 ushorts via D# pad fields (pad_interval=32 DWORDs, pad_amount=4 DWORDs).
__device__ __forceinline__ void tdm_load_tile_32x64(unsigned lds_byte_addr,
                                                    const unsigned short* gptr,
                                                    unsigned row_stride_elems)
{
    unsigned long long ga = (unsigned long long)(uintptr_t)gptr;
    v4u g0;
    g0[0] = 1u;                                       // count=1, user mode
    g0[1] = lds_byte_addr;                            // lds_addr
    g0[2] = (unsigned)(ga & 0xffffffffu);             // global_addr[31:0]
    g0[3] = (unsigned)((ga >> 32) & 0x1ffffffu)       // global_addr[56:32]
          | (2u << 30);                               // type=2 ("image")
    v8u g1;
    g1[0] = (1u << 16)                                // data_size = 2 bytes
          | (1u << 20)                                // pad_enable
          | (4u << 22)                                // pad_interval: 32 DWORDs (128B)
          | (3u << 25);                               // pad_amount: 4 DWORDs (16B)
    g1[1] = (64u << 16);                              // tensor_dim0[15:0] = 64
    g1[2] = (32u << 16);                              // tensor_dim0[31:16]=0, tensor_dim1[15:0]=32
    g1[3] = (64u << 16);                              // tensor_dim1[31:16]=0, tile_dim0=64
    g1[4] = 32u;                                      // tile_dim1=32, tile_dim2=0
    g1[5] = row_stride_elems;                         // tensor_dim0_stride[31:0]
    g1[6] = 0u;                                       // stride[47:32], dim1_stride[15:0]
    g1[7] = 0u;
    v4u gz = (v4u)(0u);
    asm volatile("tensor_load_to_lds %0, %1, %2, %3"
                 :: "s"(g0), "s"(g1), "s"(gz), "s"(gz)
                 : "memory");
}
#endif

// ---------------------------------------------------------------------------
// GEMM: C[M,N] = A[M,K] * B[K,N].  A fp32 or bf16 (row-major), B fp32.
// Block: 128 threads (4 waves), 64x64 C tile; wave = 32x32 (2x2 WMMA accs).
// B tile stored transposed in LDS so B-fragment reads are contiguous b128s.
// ---------------------------------------------------------------------------
template <bool A_BF16, bool OUT_BF16>
__global__ __launch_bounds__(128)
void gemm_bf16_wmma(const void* __restrict__ Aptr, const float* __restrict__ B,
                    void* __restrict__ Cptr, int M, int N, int K)
{
    __shared__ unsigned short As [64][34];   // rows x K          (padded)
    __shared__ unsigned short Bst[64][34];   // cols x K (transposed, padded)

    const int tid  = threadIdx.x;
    const int lane = tid & 31;
    const int wid  = tid >> 5;
    const int wr   = (wid >> 1) * 32;
    const int wc   = (wid & 1) * 32;
    const int hi   = lane >> 4;
    const int l15  = lane & 15;
    const int bm   = blockIdx.y * 64;
    const int bn   = blockIdx.x * 64;

    v8f acc[2][2] = {};

    for (int k0 = 0; k0 < K; k0 += 32) {
        // stage A tile (64 rows x 32 K) as bf16
        #pragma unroll
        for (int i = 0; i < 16; ++i) {
            int e = tid * 16 + i;
            int r = e >> 5, c = e & 31;
            unsigned short us;
            if (A_BF16) us = ((const unsigned short*)Aptr)[(size_t)(bm + r) * K + k0 + c];
            else        us = f2bf(((const float*)Aptr)[(size_t)(bm + r) * K + k0 + c]);
            As[r][c] = us;
        }
        // stage B tile (32 K x 64 cols) transposed as bf16
        #pragma unroll
        for (int i = 0; i < 16; ++i) {
            int e = tid * 16 + i;
            int r = e >> 6, c = e & 63;
            Bst[c][r] = f2bf(B[(size_t)(k0 + r) * N + bn + c]);
        }
        __syncthreads();

        // A fragments: lane L -> row (L&15), K = 16*(L>>4)+[0,16)  (contiguous)
        v16us au[2];
        #pragma unroll
        for (int mt = 0; mt < 2; ++mt) {
            int m  = wr + mt * 16 + l15;
            int kb = hi * 16;
            v16us t;
            #pragma unroll
            for (int j = 0; j < 8; ++j) {
                int k = kb + 2 * j;
                t[2 * j]     = As[m][k];
                t[2 * j + 1] = As[m][k + 1];
            }
            au[mt] = t;
        }
        // B fragments: lane L -> col (L&15), K = 16*(L>>4)+[0,16)  (contiguous)
        v16us bu[2];
        #pragma unroll
        for (int nt = 0; nt < 2; ++nt) {
            int n  = wc + nt * 16 + l15;
            int kb = hi * 16;
            v16us t;
            #pragma unroll
            for (int j = 0; j < 16; ++j) t[j] = Bst[n][kb + j];
            bu[nt] = t;
        }
        #pragma unroll
        for (int mt = 0; mt < 2; ++mt)
            #pragma unroll
            for (int nt = 0; nt < 2; ++nt)
                acc[mt][nt] = wmma_bf16(au[mt], bu[nt], acc[mt][nt]);
        __syncthreads();
    }

    #pragma unroll
    for (int mt = 0; mt < 2; ++mt)
        #pragma unroll
        for (int nt = 0; nt < 2; ++nt)
            #pragma unroll
            for (int v = 0; v < 8; ++v) {
                int row = bm + wr + mt * 16 + v + 8 * hi;
                int col = bn + wc + nt * 16 + l15;
                float val = acc[mt][nt][v];
                if (OUT_BF16) ((unsigned short*)Cptr)[(size_t)row * N + col] = f2bf(val);
                else          ((float*)Cptr)[(size_t)row * N + col] = val;
            }
}

// ---------------------------------------------------------------------------
// RoPE (interleaved even/odd pairs), bf16 in/out, fp32 math. buf: (S, heads, 64)
// ---------------------------------------------------------------------------
__global__ void rope_kernel(unsigned short* __restrict__ buf, int heads,
                            const float* __restrict__ sinT, const float* __restrict__ cosT,
                            const int* __restrict__ startp, int S)
{
    int idx = blockIdx.x * blockDim.x + threadIdx.x;
    int total = S * heads * 32;
    if (idx >= total) return;
    int i = idx & 31;
    int t = idx >> 5;
    int h = t % heads;
    int s = t / heads;
    int start0 = *startp;
    float sn = sinT[(size_t)(start0 + s) * 32 + i];
    float cs = cosT[(size_t)(start0 + s) * 32 + i];
    size_t base = ((size_t)s * heads + h) * 64;
    float e = bf2f(buf[base + 2 * i]);
    float o = bf2f(buf[base + 2 * i + 1]);
    buf[base + 2 * i]     = f2bf(cs * e - sn * o);
    buf[base + 2 * i + 1] = f2bf(sn * e + cs * o);
}

// ---------------------------------------------------------------------------
// Flash attention (causal, GQA 16Q/4KV, d=64), online softmax in WMMA C layout.
// grid = (S/64, 16 heads), block = 128 threads (4 waves, 16 q-rows each).
// K tile staged by the Tensor Data Mover (wave 0); V staged transposed in LDS.
// ---------------------------------------------------------------------------
__global__ __launch_bounds__(128)
void flash_attn(const unsigned short* __restrict__ Q,
                const unsigned short* __restrict__ Kb,
                const unsigned short* __restrict__ Vb,
                const int* __restrict__ amask,
                const int* __restrict__ startp,
                unsigned short* __restrict__ ctx)
{
    constexpr int S = 4096;
    __shared__ unsigned short Kt[32][72];      // kv rows x 64 dims (TDM-padded)
    __shared__ unsigned short Vt[64][40];      // dims x 32 kv rows (transposed)
    __shared__ unsigned short Pt[4][16][36];   // per-wave 16x32 P tile (padded)

    const int tid  = threadIdx.x;
    const int lane = tid & 31;
    const int wid  = tid >> 5;
    const int hi   = lane >> 4;
    const int l15  = lane & 15;
    const int h    = blockIdx.y;
    const int kvh  = h >> 2;
    const int qBase = blockIdx.x * 64;
    const int start0 = *startp;

    // Q fragments for this wave's 16 rows (two 16x32 A-frags cover d=64)
    v16us qu[2];
    {
        int qrow = qBase + wid * 16 + l15;
        size_t base = ((size_t)qrow * 16 + h) * 64;
        #pragma unroll
        for (int f = 0; f < 2; ++f) {
            int kb = hi * 16;
            v16us t;
            #pragma unroll
            for (int j = 0; j < 8; ++j) {
                int k = f * 32 + kb + 2 * j;
                t[2 * j]     = Q[base + k];
                t[2 * j + 1] = Q[base + k + 1];
            }
            qu[f] = t;
        }
    }

    float rmax[8], rsum[8];
    v8f o[4] = {};
    #pragma unroll
    for (int v = 0; v < 8; ++v) { rmax[v] = -3.0e38f; rsum[v] = 0.0f; }

    int nkt = (qBase + 63) / 32 + 1;
    if (nkt > S / 32) nkt = S / 32;

    for (int kt = 0; kt < nkt; ++kt) {
        // ---- stage K tile: Tensor Data Mover (one wave issues; EXEC ignored)
#if USE_TDM
        if (wid == 0) {
            const unsigned short* gK = Kb + ((size_t)(kt * 32) * 4 + kvh) * 64;
            tdm_load_tile_32x64((unsigned)(uintptr_t)&Kt[0][0], gK, /*row stride*/ 4 * 64);
            __builtin_amdgcn_s_wait_tensorcnt(0);
        }
#else
        for (int i = 0; i < 4; ++i) {
            int e = tid * 4 + i;
            int r = e >> 6, c = e & 63;
            Kt[r][c] = Kb[((size_t)(kt * 32 + r) * 4 + kvh) * 64 + c];
        }
#endif
        // ---- stage V tile transposed (all threads)
        #pragma unroll
        for (int i = 0; i < 16; ++i) {
            int e = tid * 16 + i;
            int r = e >> 6, c = e & 63;
            Vt[c][r] = Vb[((size_t)(kt * 32 + r) * 4 + kvh) * 64 + c];
        }
        __syncthreads();

        // ---- S = Q * K^T  (two 16-col tiles, each contraction d=64 -> 2 WMMAs)
        float s0[8], s1[8];
        #pragma unroll
        for (int nt = 0; nt < 2; ++nt) {
            v8f s = {};
            #pragma unroll
            for (int half = 0; half < 2; ++half) {
                v16us bu;
                int n  = nt * 16 + l15;          // kv row inside tile (B col)
                int kb = half * 32 + hi * 16;    // head-dim contraction base
                #pragma unroll
                for (int j = 0; j < 16; ++j) bu[j] = Kt[n][kb + j];  // contiguous
                s = wmma_bf16(qu[half], bu, s);
            }
            int col = kt * 32 + nt * 16 + l15;
            int am  = amask[col];
            float* dst = nt ? s1 : s0;
            #pragma unroll
            for (int v = 0; v < 8; ++v) {
                int q = qBase + wid * 16 + v + 8 * hi;
                float val = s[v] * 0.125f;       // 1/sqrt(64)
                bool ok = (col <= q + start0) && (am != 0);
                dst[v] = ok ? val : -1.0e6f;
            }
        }

        // ---- online softmax (16-lane butterflies stay inside each half-wave)
        #pragma unroll
        for (int v = 0; v < 8; ++v) {
            float t = fmaxf(s0[v], s1[v]);
            for (int off = 1; off < 16; off <<= 1) t = fmaxf(t, __shfl_xor(t, off, 32));
            float nm = fmaxf(rmax[v], t);
            float sc = __expf(rmax[v] - nm);
            rmax[v] = nm;
            float p0 = __expf(s0[v] - nm);
            float p1 = __expf(s1[v] - nm);
            float ds = p0 + p1;
            for (int off = 1; off < 16; off <<= 1) ds += __shfl_xor(ds, off, 32);
            rsum[v] = rsum[v] * sc + ds;
            #pragma unroll
            for (int j = 0; j < 4; ++j) o[j][v] *= sc;
            int m = v + 8 * hi;
            Pt[wid][m][l15]      = f2bf(p0);
            Pt[wid][m][16 + l15] = f2bf(p1);
        }
        __syncthreads();

        // ---- O += P * V   (P: 16x32 A-frag; V: four 16-col B-frags, contiguous)
        v16us pu;
        {
            int kb = hi * 16;
            #pragma unroll
            for (int j = 0; j < 8; ++j) {
                int k = kb + 2 * j;
                pu[2 * j]     = Pt[wid][l15][k];
                pu[2 * j + 1] = Pt[wid][l15][k + 1];
            }
        }
        #pragma unroll
        for (int j = 0; j < 4; ++j) {
            v16us vu;
            int n  = j * 16 + l15;
            int kb = hi * 16;
            #pragma unroll
            for (int t2 = 0; t2 < 16; ++t2) vu[t2] = Vt[n][kb + t2];  // contiguous
            o[j] = wmma_bf16(pu, vu, o[j]);
        }
        __syncthreads();
    }

    // normalize, apply q-mask, store ctx (bf16, layout (S,16,64))
    #pragma unroll
    for (int v = 0; v < 8; ++v) {
        int q = qBase + wid * 16 + v + 8 * hi;
        float inv = 1.0f / rsum[v];
        float qm  = (amask[q] != 0) ? 1.0f : 0.0f;
        #pragma unroll
        for (int j = 0; j < 4; ++j) {
            float val = o[j][v] * inv * qm;
            ctx[((size_t)q * 16 + h) * 64 + j * 16 + l15] = f2bf(val);
        }
    }
}

// ---------------------------------------------------------------------------
extern "C" void kernel_launch(void* const* d_in, const int* in_sizes, int n_in,
                              void* d_out, int out_size, void* d_ws, size_t ws_size,
                              hipStream_t stream)
{
    (void)in_sizes; (void)n_in; (void)out_size; (void)ws_size;
    const float* x     = (const float*)d_in[0];
    const int*   amask = (const int*)d_in[1];
    const int*   start = (const int*)d_in[2];
    const float* wq    = (const float*)d_in[3];
    const float* wk    = (const float*)d_in[4];
    const float* wv    = (const float*)d_in[5];
    const float* wo    = (const float*)d_in[6];
    const float* sinT  = (const float*)d_in[7];
    const float* cosT  = (const float*)d_in[8];
    float* out = (float*)d_out;

    const int S = 4096, H = 1024, KV = 256;

    unsigned short* qbuf = (unsigned short*)d_ws;         // 4096x1024 bf16
    unsigned short* kbuf = qbuf + (size_t)S * H;          // 4096x256 bf16
    unsigned short* vbuf = kbuf + (size_t)S * KV;         // 4096x256 bf16
    unsigned short* ctx  = vbuf + (size_t)S * KV;         // 4096x1024 bf16

    dim3 blk(128);
    gemm_bf16_wmma<false, true><<<dim3(H  / 64, S / 64), blk, 0, stream>>>(x, wq, qbuf, S, H,  H);
    gemm_bf16_wmma<false, true><<<dim3(KV / 64, S / 64), blk, 0, stream>>>(x, wk, kbuf, S, KV, H);
    gemm_bf16_wmma<false, true><<<dim3(KV / 64, S / 64), blk, 0, stream>>>(x, wv, vbuf, S, KV, H);
    int qp = S * 16 * 32, kp = S * 4 * 32;
    rope_kernel<<<(qp + 255) / 256, 256, 0, stream>>>(qbuf, 16, sinT, cosT, start, S);
    rope_kernel<<<(kp + 255) / 256, 256, 0, stream>>>(kbuf, 4,  sinT, cosT, start, S);
    flash_attn<<<dim3(S / 64, 16), blk, 0, stream>>>(qbuf, kbuf, vbuf, amask, start, ctx);
    gemm_bf16_wmma<true, false><<<dim3(H / 64, S / 64), blk, 0, stream>>>(ctx, wo, out, S, H, H);
}
